// BoundaryEnhancedLoss_12463995093244
// MI455X (gfx1250) — matrix operand order repaired and checked by reading, not compiled, not run on hardware
//
#include <hip/hip_runtime.h>
#include <hip/hip_bf16.h>

typedef __attribute__((ext_vector_type(16))) _Float16 v16h;
typedef __attribute__((ext_vector_type(8)))  float    v8f;

#define IMG      512
#define NB       32
#define NPIX     ((float)(NB * IMG * IMG))
#define TILE     16
#define HALO     2
#define TW       (TILE + 2 * HALO)   // 20
#define TSTRIDE  22                  // even (8B-aligned rows) + bank-conflict-free halves
#define WAVES_PB 8

// Accumulator layout in d_ws (floats):
// [0]=ce_sum  [1]=valid_cnt  [2]=focal_sum
// [3..34]=inter_b  [35..66]=bp_b  [67..98]=bt_b
#define ACC_N 99

__global__ void bel_init(float* acc) {
    int t = threadIdx.x;
    if (t < ACC_N) acc[t] = 0.0f;
}

__global__ __launch_bounds__(256) void bel_main(const float* __restrict__ pred,
                                                const int*   __restrict__ target,
                                                float* __restrict__ acc) {
    const int lane = threadIdx.x & 31;
    const int wv   = threadIdx.x >> 5;
    const int gw   = blockIdx.x * WAVES_PB + wv;     // one 16x16 tile per wave
    const int img  = gw >> 10;                       // 1024 tiles per image
    const int tin  = gw & 1023;
    const int ty   = (tin >> 5) << 4;
    const int tx   = (tin & 31) << 4;

    __shared__ int T[WAVES_PB][TW][TSTRIDE];         // 20x20 halo tile per wave

    const int* tgt = target + (size_t)img * (IMG * IMG);

    const bool interior = (ty > 0) && (ty < IMG - TILE) && (tx > 0) && (tx < IMG - TILE);
    if (interior) {
        // CDNA5 async global->LDS copy (ASYNCcnt-tracked), unpadded interior halo.
        // 20 rows x 10 b64 pairs = 200 lane-issues.
        const int* base = tgt + (ty - HALO) * IMG + (tx - HALO);   // wave-uniform
        for (int k = 0; k < 7; ++k) {
            int idx = lane + k * 32;
            if (idx < 200) {
                int row = idx / 10;
                int c2  = (idx % 10) * 2;
                unsigned lds_off = (unsigned)(size_t)&T[wv][row][c2];
                int goff = (row * IMG + c2) * 4;
                asm volatile("global_load_async_to_lds_b64 %0, %1, %2"
                             :
                             : "v"(lds_off), "v"(goff), "s"(base)
                             : "memory");
            }
        }
        asm volatile("s_wait_asynccnt 0x0" ::: "memory");
    } else {
        // Edge tiles: scalar path with zero padding ('SAME' zero-pad conv)
        for (int i = lane; i < TW * TW; i += 32) {
            int rr = i / TW, cc = i % TW;
            int gy = ty + rr - HALO, gx = tx + cc - HALO;
            int v = 0;
            if (gy >= 0 && gy < IMG && gx >= 0 && gx < IMG) v = tgt[gy * IMG + gx];
            T[wv][rr][cc] = v;
        }
    }
    __syncthreads();

    const bool hi = (lane >= 16);
    const int  n  = lane & 15;   // column for B/D
    const int  m  = lane & 15;   // A-matrix row held by this lane

    // A: 16x32 band matrix, A[m][k] = 1 for m <= k <= m+4  (exact in f16)
    // B: 32x16, B[k][n] = horizontal 5-sum of target row (ty+k-2), rows >=20 are 0.
    // Contiguous k = 16*h + j for both A and B (lanes 0-15 k 0-15, lanes 16-31 k 16-31), not the ISA table order.
    v16h a, b;
#pragma unroll
    for (int j = 0; j < 16; ++j) {
        int ka = j + (hi ? 16 : 0);
        a[j] = (_Float16)((ka >= m && ka <= m + 4) ? 1.0f : 0.0f);
        int kb = j + (hi ? 16 : 0);
        float rs = 0.0f;
        if (kb < TW) {
            int s = 0;
#pragma unroll
            for (int dx = 0; dx < 5; ++dx) s += T[wv][kb][n + dx];
            rs = (float)s;   // 0..5, exact in f16
        }
        b[j] = (_Float16)rs;
    }

    v8f c = {};
    // S(16x16) = band(16x32) x Rsum(32x16): vertical 5-tap of horizontal 5-sums = 5x5 box sum
    c = __builtin_amdgcn_wmma_f32_16x16x32_f16(false, a, false, b,
                                               (short)0, c, false, false);

    const float* p0 = pred + (size_t)(img * 2 + 0) * (IMG * IMG);
    const float* p1 = pred + (size_t)(img * 2 + 1) * (IMG * IMG);

    float ce_s = 0.f, cnt_s = 0.f, foc_s = 0.f, bp_s = 0.f, bt_s = 0.f, in_s = 0.f;

#pragma unroll
    for (int r = 0; r < 8; ++r) {
        int   my = r + (hi ? 8 : 0);          // D layout: VGPR r -> rows r / r+8
        int   y = ty + my, x = tx + n;
        float s = c[r];                        // exact integer 0..25
        float bnd = (s > 0.5f && s < 24.5f) ? 1.0f : 0.0f;   // dilated - eroded

        int t = T[wv][my + HALO][n + HALO];
        size_t off = (size_t)y * IMG + x;
        float a0 = p0[off], a1 = p1[off];

        float mx  = fmaxf(a0, a1);
        float lse = mx + __logf(__expf(a0 - mx) + __expf(a1 - mx));
        float sel = t ? a1 : a0;
        float ce  = lse - sel;                 // -log p[target]
        float valid = (t != 250) ? 1.0f : 0.0f;

        float pt = __expf(-ce);
        float om = 1.0f - pt;
        foc_s += 0.25f * om * om * ce;         // focal over ALL pixels
        ce_s  += ce * valid;
        cnt_s += valid;

        float prob1 = __expf(a1 - lse);
        float bp = prob1 * bnd;
        float bt = (float)t * bnd;
        bp_s += bp; bt_s += bt; in_s += bp * bt;
    }

    // wave32 tree reduction
#pragma unroll
    for (int o = 16; o > 0; o >>= 1) {
        ce_s  += __shfl_down(ce_s,  o, 32);
        cnt_s += __shfl_down(cnt_s, o, 32);
        foc_s += __shfl_down(foc_s, o, 32);
        bp_s  += __shfl_down(bp_s,  o, 32);
        bt_s  += __shfl_down(bt_s,  o, 32);
        in_s  += __shfl_down(in_s,  o, 32);
    }
    if (lane == 0) {
        atomicAdd(&acc[0], ce_s);
        atomicAdd(&acc[1], cnt_s);
        atomicAdd(&acc[2], foc_s);
        atomicAdd(&acc[3  + img], in_s);
        atomicAdd(&acc[35 + img], bp_s);
        atomicAdd(&acc[67 + img], bt_s);
    }
}

__global__ void bel_finalize(const float* __restrict__ acc, float* __restrict__ out) {
    int b = threadIdx.x;   // 32 threads, one per image
    float dice = 0.0f;
    if (b < NB) {
        float inter = acc[3 + b];
        float uni   = acc[35 + b] + acc[67 + b];
        dice = 2.0f * inter / (uni + 1e-8f);
    }
#pragma unroll
    for (int o = 16; o > 0; o >>= 1) dice += __shfl_down(dice, o, 32);
    if (b == 0) {
        float ce    = acc[0] / fmaxf(acc[1], 1.0f);
        float focal = acc[2] / NPIX;
        float bdice = 1.0f - dice / (float)NB;
        out[0] = ce + focal + bdice;
    }
}

extern "C" void kernel_launch(void* const* d_in, const int* in_sizes, int n_in,
                              void* d_out, int out_size, void* d_ws, size_t ws_size,
                              hipStream_t stream) {
    const float* pred   = (const float*)d_in[0];
    const int*   target = (const int*)d_in[1];     // jax int64 canonicalizes to int32
    float*       acc    = (float*)d_ws;
    float*       out    = (float*)d_out;

    (void)in_sizes; (void)n_in; (void)out_size; (void)ws_size;

    bel_init<<<1, 128, 0, stream>>>(acc);
    // 32 images * 1024 tiles = 32768 waves; 8 waves/block -> 4096 blocks
    bel_main<<<4096, 256, 0, stream>>>(pred, target, acc);
    bel_finalize<<<1, 32, 0, stream>>>(acc, out);
}
